// NeighborInteraction_11622181503055
// MI455X (gfx1250) — matrix operand order repaired and checked by reading, not compiled, not run on hardware
//
#include <hip/hip_runtime.h>

#define NN   8192
#define IND  16
#define HD   128
#define OUTD 16

typedef __attribute__((ext_vector_type(16))) __bf16 v16bf;
typedef __attribute__((ext_vector_type(8)))  float  v8f;

static __device__ __forceinline__ unsigned short f2bf(float f) {
    unsigned u = __builtin_bit_cast(unsigned, f);
    u += 0x7fffu + ((u >> 16) & 1u);          // round-to-nearest-even
    return (unsigned short)(u >> 16);
}

// ---------------------------------------------------------------------------
// Kernel A: two 3-layer MLPs per row. One wave per row; lane computes units
// u = lane + 32p (p=0..3). Hidden activations shared through per-wave LDS
// buffers (same-wave DS ops are in-order on CDNA5, no barrier needed).
// y1 stored as bf16 (feeds WMMA directly), y2 as f32 (epilogue multiply).
// ---------------------------------------------------------------------------
static __device__ __forceinline__ void mlp3(
    const float* __restrict__ xr, float* t0, float* t1,
    const float* __restrict__ w1, const float* __restrict__ b1,
    const float* __restrict__ w2, const float* __restrict__ b2,
    const float* __restrict__ w3, const float* __restrict__ b3,
    int lane, float yout[4])
{
    #pragma unroll
    for (int p = 0; p < 4; ++p) {
        const int u = lane + 32 * p;
        float a = b1[u];
        const float* wr = &w1[u * IND];
        #pragma unroll
        for (int k = 0; k < IND; ++k) a = fmaf(xr[k], wr[k], a);
        t0[u] = tanhf(a);
    }
    #pragma unroll
    for (int p = 0; p < 4; ++p) {
        const int u = lane + 32 * p;
        float a = b2[u];
        const float* wr = &w2[u * HD];
        for (int k = 0; k < HD; ++k) a = fmaf(t0[k], wr[k], a);
        t1[u] = tanhf(a);
    }
    #pragma unroll
    for (int p = 0; p < 4; ++p) {
        const int u = lane + 32 * p;
        float a = b3[u];
        const float* wr = &w3[u * HD];
        for (int k = 0; k < HD; ++k) a = fmaf(t1[k], wr[k], a);
        yout[p] = a;
    }
}

__global__ __launch_bounds__(256) void mlp_kernel(
    const float* __restrict__ x,
    const float* __restrict__ w11, const float* __restrict__ b11,
    const float* __restrict__ w12, const float* __restrict__ b12,
    const float* __restrict__ w13, const float* __restrict__ b13,
    const float* __restrict__ w21, const float* __restrict__ b21,
    const float* __restrict__ w22, const float* __restrict__ b22,
    const float* __restrict__ w23, const float* __restrict__ b23,
    unsigned short* __restrict__ y1bf, float* __restrict__ y2f)
{
    __shared__ float xb[8][IND];
    __shared__ float buf[8][2][HD];
    const int lane = threadIdx.x & 31;
    const int wave = threadIdx.x >> 5;
    const int wgid = blockIdx.x * 8 + wave;
    const int nw   = gridDim.x * 8;

    for (int row = wgid; row < NN; row += nw) {
        if (lane < IND) xb[wave][lane] = x[row * IND + lane];
        float y1v[4], y2v[4];
        mlp3(xb[wave], buf[wave][0], buf[wave][1],
             w11, b11, w12, b12, w13, b13, lane, y1v);
        mlp3(xb[wave], buf[wave][0], buf[wave][1],
             w21, b21, w22, b22, w23, b23, lane, y2v);
        #pragma unroll
        for (int p = 0; p < 4; ++p) {
            y1bf[(size_t)row * HD + lane + 32 * p] = f2bf(y1v[p]);
            y2f [(size_t)row * HD + lane + 32 * p] = y2v[p];
        }
    }
}

// ---------------------------------------------------------------------------
// Kernel B: s[j,h] = y2[j,h] * sum_i adj[i,j] * y1[i,h]  via bf16 WMMA.
// Block = 4 waves, 64 j-rows; wave w owns j-strip w (16 rows) x full H=128
// -> 8 accumulator tiles. Double-buffered LDS (one barrier/iter) + register
// prefetch of the next K-chunk so VMEM latency overlaps WMMA execution.
// LDS tiles are i-minor with 40-ushort (80B) row pitch: every fragment read
// is a 16B-aligned ds_load_b128 on a conflict-free 20-bank stride.
// ---------------------------------------------------------------------------
__global__ __launch_bounds__(128) void gemm_kernel(
    const float* __restrict__ adj,
    const unsigned short* __restrict__ y1bf,
    const float* __restrict__ y2f,
    float* __restrict__ s_ws)
{
    __shared__ unsigned short ldsA[2][64 * 40];   // adj^T tile: [j_loc][i_loc]
    __shared__ unsigned short ldsY[2][128 * 40];  // y1^T  tile: [h][i_loc]

    const int tid  = threadIdx.x;
    const int lane = tid & 31;
    const int wave = tid >> 5;
    const int jb   = blockIdx.x * 64;

    v8f acc[8] = {};

    const int c4 = (tid & 15) * 4;   // adj staging: j_loc quad
    const int ib = tid >> 4;         // staging i base (0..7)
    const int h8 = (tid & 15) * 8;   // y1 staging: h octet

    float4 ga[4];                    // prefetched adj chunk
    uint4  gy[4];                    // prefetched y1 chunk

    auto load_global = [&](int k0) {
        #pragma unroll
        for (int s = 0; s < 4; ++s) {
            const int i = ib + 8 * s;
            ga[s] = *(const float4*)__builtin_assume_aligned(
                        &adj[(size_t)(k0 + i) * NN + jb + c4], 16);
            gy[s] = *(const uint4*)__builtin_assume_aligned(
                        &y1bf[(size_t)(k0 + i) * HD + h8], 16);
        }
    };
    auto store_lds = [&](int b) {
        #pragma unroll
        for (int s = 0; s < 4; ++s) {
            const int i = ib + 8 * s;
            ldsA[b][(c4 + 0) * 40 + i] = f2bf(ga[s].x);
            ldsA[b][(c4 + 1) * 40 + i] = f2bf(ga[s].y);
            ldsA[b][(c4 + 2) * 40 + i] = f2bf(ga[s].z);
            ldsA[b][(c4 + 3) * 40 + i] = f2bf(ga[s].w);
            const unsigned short* pv = (const unsigned short*)&gy[s];
            #pragma unroll
            for (int q = 0; q < 8; ++q) ldsY[b][(h8 + q) * 40 + i] = pv[q];
        }
    };

    load_global(0);
    store_lds(0);
    load_global(32);

    // A fragment address pieces (documented 16-bit A layout):
    //   lanes 0-15 : elems 0..7 = K0..7,  elems 8..15 = K16..23
    //   lanes16-31 : elems 0..7 = K8..15, elems 8..15 = K24..31
    const int jr = wave * 16 + (lane & 15);
    const int ab = (lane < 16) ? 0 : 8;
    const int bb = (lane < 16) ? 0 : 8;

    for (int k0 = 0, it = 0; k0 < NN; k0 += 32, ++it) {
        const int cur = it & 1, nxt = cur ^ 1;
        __syncthreads();

        // fragments for this chunk
        union { uint4 q[2]; v16bf v; } A;
        A.q[0] = *(const uint4*)&ldsA[cur][jr * 40 + ab];
        A.q[1] = *(const uint4*)&ldsA[cur][jr * 40 + ab + 16];
        union { uint4 q[2]; v16bf v; } B[8];
        #pragma unroll
        for (int t = 0; t < 8; ++t) {
            const int hr = t * 16 + (lane & 15);
            B[t].q[0] = *(const uint4*)&ldsY[cur][hr * 40 + bb];
            B[t].q[1] = *(const uint4*)&ldsY[cur][hr * 40 + bb + 16];
        }

        // stage next chunk into the other buffer (overlaps WMMA execution)
        if (k0 + 32 < NN) store_lds(nxt);

        #pragma unroll
        for (int t = 0; t < 8; ++t)
            acc[t] = __builtin_amdgcn_wmma_f32_16x16x32_bf16(
                false, A.v, false, B[t].v, (short)0, acc[t], false, false);

        // issue global loads for chunk k0+64 (land during next iteration)
        if (k0 + 64 < NN) load_global(k0 + 64);
    }

    // ---- epilogue: fuse y2 multiply, store s ----
    // D tile layout: VGPR r -> M = r (+8 for lanes 16-31), N = lane&15.
    const int j0   = jb + wave * 16;
    const int jofs = (lane < 16) ? 0 : 8;
    #pragma unroll
    for (int t = 0; t < 8; ++t) {
        const int h = t * 16 + (lane & 15);
        #pragma unroll
        for (int r = 0; r < 8; ++r) {
            const int j = j0 + r + jofs;
            s_ws[(size_t)j * HD + h] = acc[t][r] * y2f[(size_t)j * HD + h];
        }
    }
}

// ---------------------------------------------------------------------------
// Kernel C: out[j,o] = bagg[o] + sum_h wagg[o,h] * s[j,h]
// ---------------------------------------------------------------------------
__global__ __launch_bounds__(256) void proj_kernel(
    const float* __restrict__ s_ws, const float* __restrict__ wagg,
    const float* __restrict__ bagg, float* __restrict__ out)
{
    __shared__ float wg[OUTD * HD];
    for (int i = threadIdx.x; i < OUTD * HD; i += 256) wg[i] = wagg[i];
    __syncthreads();

    const int jl = threadIdx.x >> 2;
    const int og = (threadIdx.x & 3) * 4;
    const int j  = blockIdx.x * 64 + jl;

    float a0 = 0.f, a1 = 0.f, a2 = 0.f, a3 = 0.f;
    const float* sr = &s_ws[(size_t)j * HD];
    for (int h = 0; h < HD; ++h) {
        const float sv = sr[h];
        a0 = fmaf(wg[(og + 0) * HD + h], sv, a0);
        a1 = fmaf(wg[(og + 1) * HD + h], sv, a1);
        a2 = fmaf(wg[(og + 2) * HD + h], sv, a2);
        a3 = fmaf(wg[(og + 3) * HD + h], sv, a3);
    }
    out[(size_t)j * OUTD + og + 0] = a0 + bagg[og + 0];
    out[(size_t)j * OUTD + og + 1] = a1 + bagg[og + 1];
    out[(size_t)j * OUTD + og + 2] = a2 + bagg[og + 2];
    out[(size_t)j * OUTD + og + 3] = a3 + bagg[og + 3];
}

// ---------------------------------------------------------------------------
extern "C" void kernel_launch(void* const* d_in, const int* in_sizes, int n_in,
                              void* d_out, int out_size, void* d_ws, size_t ws_size,
                              hipStream_t stream) {
    (void)in_sizes; (void)n_in; (void)out_size; (void)ws_size;
    // dict order: t, x, adj, w11,b11, w12,b12, w13,b13, w21,b21, w22,b22, w23,b23, wagg,bagg
    const float* x    = (const float*)d_in[1];
    const float* adj  = (const float*)d_in[2];
    const float* w11  = (const float*)d_in[3];
    const float* b11  = (const float*)d_in[4];
    const float* w12  = (const float*)d_in[5];
    const float* b12  = (const float*)d_in[6];
    const float* w13  = (const float*)d_in[7];
    const float* b13  = (const float*)d_in[8];
    const float* w21  = (const float*)d_in[9];
    const float* b21  = (const float*)d_in[10];
    const float* w22  = (const float*)d_in[11];
    const float* b22  = (const float*)d_in[12];
    const float* w23  = (const float*)d_in[13];
    const float* b23  = (const float*)d_in[14];
    const float* wagg = (const float*)d_in[15];
    const float* bagg = (const float*)d_in[16];
    float* out = (float*)d_out;

    // workspace: y1 bf16 (2MB) | y2 f32 (4MB) | s f32 (4MB)
    unsigned short* y1bf = (unsigned short*)d_ws;
    float* y2f  = (float*)((char*)d_ws + (size_t)2 * 1024 * 1024);
    float* s_ws = (float*)((char*)d_ws + (size_t)6 * 1024 * 1024);

    mlp_kernel<<<256, 256, 0, stream>>>(x, w11, b11, w12, b12, w13, b13,
                                        w21, b21, w22, b22, w23, b23, y1bf, y2f);
    gemm_kernel<<<NN / 64, 128, 0, stream>>>(adj, y1bf, y2f, s_ws);
    proj_kernel<<<NN / 64, 256, 0, stream>>>(s_ws, wagg, bagg, out);
}
